// LayerRouterModel_8598524526995
// MI455X (gfx1250) — matrix operand order repaired and checked by reading, not compiled, not run on hardware
//
#include <hip/hip_runtime.h>

typedef __attribute__((ext_vector_type(16))) __bf16 bf16x16;
typedef __attribute__((ext_vector_type(8)))  float  f32x8;

union FragU { bf16x16 f; uint4 u[2]; };

constexpr int B_  = 2;
constexpr int S_  = 2048;
constexpr int H_  = 2048;
constexpr int NH_ = 16;
constexpr int NKV_= 8;
constexpr int HD_ = 128;

__device__ __forceinline__ __bf16 f2bf(float f) {
  union { float f; unsigned u; } v; v.f = f;
  unsigned r = (v.u + 0x7FFFu + ((v.u >> 16) & 1u)) >> 16;
  union { unsigned short s; __bf16 b; } o; o.s = (unsigned short)r; return o.b;
}
__device__ __forceinline__ float bf2f(__bf16 b) {
  union { unsigned short s; __bf16 b; } i; i.b = b;
  union { float f; unsigned u; } v; v.u = ((unsigned)i.s) << 16; return v.f;
}
__device__ __forceinline__ f32x8 wmma_bf16(bf16x16 a, bf16x16 b, f32x8 c) {
  return __builtin_amdgcn_wmma_f32_16x16x32_bf16(false, a, false, b, (short)0, c, false, false);
}

// CDNA5 async global->LDS copy (16B per lane), tracked on ASYNCcnt.
// vdst = LDS byte offset (low 32 bits of generic shared pointer).
__device__ __forceinline__ void async_ld_b128(void* lds, const void* gaddr) {
  unsigned off = (unsigned)(size_t)lds;
  asm volatile("global_load_async_to_lds_b128 %0, %1, off"
               :: "v"(off), "v"(gaddr) : "memory");
}
__device__ __forceinline__ void wait_async0() {
  asm volatile("s_wait_asynccnt 0x0" ::: "memory");
}

// ---------------------------------------------------------------------------
// fp32 -> bf16 pack with optional column placement into a wider matrix
// ---------------------------------------------------------------------------
__global__ void pack_bf16_kernel(const float* __restrict__ src, __bf16* __restrict__ dst,
                                 long long K, long long N, long long dstLd, long long colOff)
{
  size_t idx = (size_t)blockIdx.x * 256 + threadIdx.x;
  size_t total = (size_t)K * N;
  if (idx >= total) return;
  size_t k = idx / (size_t)N;
  size_t n = idx - k * (size_t)N;
  dst[k * (size_t)dstLd + (size_t)colOff + n] = f2bf(src[idx]);
}

// ---------------------------------------------------------------------------
// RMSNorm(emb)*in_w ++ RMSNorm(hidden)*hn_w  -> x bf16 (4096 wide)
// ---------------------------------------------------------------------------
__global__ __launch_bounds__(256) void rms_concat_kernel(
    const float* __restrict__ emb, const float* __restrict__ hid,
    const float* __restrict__ in_w, const float* __restrict__ hn_w,
    __bf16* __restrict__ X)
{
  __shared__ float red[2][256];
  const int row = blockIdx.x;
  const int tid = threadIdx.x;
  const float* e = emb + (size_t)row * H_;
  const float* h = hid + (size_t)row * H_;
  float se = 0.f, sh = 0.f;
  for (int i = tid; i < H_; i += 256) { float a = e[i]; se += a * a; float b = h[i]; sh += b * b; }
  red[0][tid] = se; red[1][tid] = sh;
  __syncthreads();
  for (int s = 128; s > 0; s >>= 1) {
    if (tid < s) { red[0][tid] += red[0][tid + s]; red[1][tid] += red[1][tid + s]; }
    __syncthreads();
  }
  float re = rsqrtf(red[0][0] / (float)H_ + 1e-6f);
  float rh = rsqrtf(red[1][0] / (float)H_ + 1e-6f);
  __bf16* x = X + (size_t)row * 4096;
  for (int i = tid; i < H_; i += 256) {
    x[i]      = f2bf(e[i] * re * in_w[i]);
    x[H_ + i] = f2bf(h[i] * rh * hn_w[i]);
  }
}

__global__ __launch_bounds__(256) void rms_post_kernel(
    const float* __restrict__ hid, const float* __restrict__ w, __bf16* __restrict__ out)
{
  __shared__ float red[256];
  const int row = blockIdx.x, tid = threadIdx.x;
  const float* x = hid + (size_t)row * H_;
  float s = 0.f;
  for (int i = tid; i < H_; i += 256) { float a = x[i]; s += a * a; }
  red[tid] = s; __syncthreads();
  for (int t = 128; t > 0; t >>= 1) { if (tid < t) red[tid] += red[tid + t]; __syncthreads(); }
  float r = rsqrtf(red[0] / (float)H_ + 1e-6f);
  __bf16* o = out + (size_t)row * H_;
  for (int i = tid; i < H_; i += 256) o[i] = f2bf(x[i] * r * w[i]);
}

// ---------------------------------------------------------------------------
// bf16 WMMA GEMM: C(f32, MxN) = A(bf16, MxK) @ B(bf16, KxN) [+ resid]
// Block tile 128x256, 8 waves (2x4), wave tile 64x64 (16 WMMA accumulators),
// K staged 32 at a time. A tile staged via async global->LDS b128; B tile
// staged transposed so B-fragments are contiguous 32B ds_load_b128 reads.
// ---------------------------------------------------------------------------
template<int MODE>
__global__ __launch_bounds__(256) void gemm_bf16_kernel(
    const __bf16* __restrict__ A, const __bf16* __restrict__ B,
    float* __restrict__ C, const float* __restrict__ resid,
    int M, int N, int K)
{
  __shared__ __bf16 Alds[128 * 48];   // [m][k], stride 48
  __shared__ __bf16 Blds[256 * 48];   // transposed [n][k], stride 48
  const int tid  = threadIdx.x;
  const int wave = tid >> 5, lane = tid & 31;
  const int lm = lane & 15, lg = lane >> 4;
  const int wm = wave >> 2, wn = wave & 3;
  const int m0 = blockIdx.y * 128, n0 = blockIdx.x * 256;

  f32x8 acc[4][4];
  for (int i = 0; i < 4; i++) for (int j = 0; j < 4; j++)
    for (int r = 0; r < 8; r++) acc[i][j][r] = 0.f;

  // A staging: 128 rows x 32 cols, 2 async b128 per thread
  const int arow  = tid >> 1;
  const int acol0 = (tid & 1) * 16;
  // B staging: 32 rows x 256 cols, transposed scatter, 32 elements per thread
  const int brow = tid >> 3, bcol = (tid & 7) * 32;

  for (int k0 = 0; k0 < K; k0 += 32) {
    { // stage A tile (async DMA to LDS)
      const __bf16* ag = A + (size_t)(m0 + arow) * K + k0 + acol0;
      __bf16* al = Alds + arow * 48 + acol0;
      async_ld_b128(al,     ag);
      async_ld_b128(al + 8, ag + 8);
    }
    { // stage B tile (transposed)
      const __bf16* bg = B + (size_t)(k0 + brow) * N + n0 + bcol;
      #pragma unroll
      for (int j = 0; j < 32; j++) Blds[(bcol + j) * 48 + brow] = bg[j];
    }
    if (k0 + 32 < K) { // prefetch next B tile -> global_prefetch_b8
      __builtin_prefetch(B + (size_t)(k0 + 32 + brow) * N + n0 + bcol, 0, 1);
    }
    wait_async0();
    __syncthreads();

    FragU af[4], bf[4];
    #pragma unroll
    for (int mi = 0; mi < 4; mi++) {
      const __bf16* p = Alds + (wm * 64 + mi * 16 + lm) * 48;
      af[mi].u[0] = *(const uint4*)(p + lg * 8);
      af[mi].u[1] = *(const uint4*)(p + 16 + lg * 8);
    }
    #pragma unroll
    for (int ni = 0; ni < 4; ni++) {
      const __bf16* p = Blds + (wn * 64 + ni * 16 + lm) * 48;
      bf[ni].u[0] = *(const uint4*)(p + lg * 8);
      bf[ni].u[1] = *(const uint4*)(p + 16 + lg * 8);
    }
    #pragma unroll
    for (int mi = 0; mi < 4; mi++)
      #pragma unroll
      for (int ni = 0; ni < 4; ni++)
        acc[mi][ni] = wmma_bf16(af[mi].f, bf[ni].f, acc[mi][ni]);
    __syncthreads();
  }

  #pragma unroll
  for (int mi = 0; mi < 4; mi++)
    #pragma unroll
    for (int ni = 0; ni < 4; ni++)
      #pragma unroll
      for (int r = 0; r < 8; r++) {
        int row = m0 + wm * 64 + mi * 16 + lg * 8 + r;
        int col = n0 + wn * 64 + ni * 16 + lm;
        size_t idx = (size_t)row * N + col;
        float v = acc[mi][ni][r];
        if (MODE == 1) v += resid[idx];
        C[idx] = v;
      }
}

// ---------------------------------------------------------------------------
// RoPE on q/k, reshape qkv f32 (row-major [b*s][4096]) into head-major bf16
// ---------------------------------------------------------------------------
__global__ __launch_bounds__(256) void rope_kernel(
    const float* __restrict__ qkv, const float* __restrict__ cosb, const float* __restrict__ sinb,
    __bf16* __restrict__ Qo, __bf16* __restrict__ Ko, __bf16* __restrict__ Vo)
{
  const int row = blockIdx.x;         // b*S + s
  const int b = row >> 11, s = row & (S_ - 1);
  const int tid = threadIdx.x;
  const float* x  = qkv  + (size_t)row * 4096;
  const float* cs = cosb + (size_t)row * HD_;
  const float* sn = sinb + (size_t)row * HD_;
  for (int n = tid; n < NH_ * HD_; n += 256) {
    int h = n >> 7, d = n & 127;
    float v  = x[n];
    float v2 = x[(h << 7) + ((d < 64) ? d + 64 : d - 64)];
    float o  = (d < 64) ? v * cs[d] - v2 * sn[d] : v * cs[d] + v2 * sn[d];
    Qo[(((size_t)b * NH_ + h) * S_ + s) * HD_ + d] = f2bf(o);
  }
  for (int n = tid; n < NKV_ * HD_; n += 256) {
    int h = n >> 7, d = n & 127;
    float v  = x[2048 + n];
    float v2 = x[2048 + (h << 7) + ((d < 64) ? d + 64 : d - 64)];
    float o  = (d < 64) ? v * cs[d] - v2 * sn[d] : v * cs[d] + v2 * sn[d];
    Ko[(((size_t)b * NKV_ + h) * S_ + s) * HD_ + d] = f2bf(o);
  }
  for (int n = tid; n < NKV_ * HD_; n += 256) {
    int h = n >> 7, d = n & 127;
    Vo[(((size_t)b * NKV_ + h) * S_ + s) * HD_ + d] = f2bf(x[3072 + n]);
  }
}

// ---------------------------------------------------------------------------
// Per-position diagonal cache logits: L[b,h,s,c] = scale * q . ks_rest[c]
// ---------------------------------------------------------------------------
__global__ __launch_bounds__(128) void diag_logits_kernel(
    const __bf16* __restrict__ Qb, const float* __restrict__ cache_k,
    const __bf16* __restrict__ Kb, float* __restrict__ L)
{
  __shared__ float sbuf[128];
  const int bhs = blockIdx.x;
  const int d = threadIdx.x;
  const int s = bhs & (S_ - 1);
  const int bh = bhs >> 11;
  const int b = bh >> 4, h = bh & 15;
  const float scale = 0.08838834764831845f;   // 1/sqrt(128)
  float qv = bf2f(Qb[(size_t)bhs * HD_ + d]);
  float kv[3];
  kv[0] = cache_k[(((size_t)(1 * B_ * NH_) + bh) * S_ + s) * HD_ + d];
  kv[1] = cache_k[(((size_t)(2 * B_ * NH_) + bh) * S_ + s) * HD_ + d];
  kv[2] = bf2f(Kb[(((size_t)b * NKV_ + (h >> 1)) * S_ + s) * HD_ + d]);
  for (int c = 0; c < 3; c++) {
    sbuf[d] = qv * kv[c];
    __syncthreads();
    for (int t = 64; t > 0; t >>= 1) { if (d < t) sbuf[d] += sbuf[d + t]; __syncthreads(); }
    if (d == 0) L[(size_t)bhs * 3 + c] = sbuf[0] * scale;
    __syncthreads();
  }
}

// ---------------------------------------------------------------------------
// Flash attention vs cache_k[0]/cache_v[0] + 3 diagonal cache columns.
// Block = one (b,h) x 128 query rows; 8 waves x 16 rows each.
// K tile staged via async global->LDS; V tile staged transposed.
// ---------------------------------------------------------------------------
__global__ __launch_bounds__(256) void attention_kernel(
    const __bf16* __restrict__ Qb, const __bf16* __restrict__ K0b,
    const __bf16* __restrict__ V0b, const float* __restrict__ logits3,
    const float* __restrict__ cache_v, const __bf16* __restrict__ Vnew,
    __bf16* __restrict__ Out)
{
  __shared__ __bf16 Klds[32 * 136];     // [key][d]
  __shared__ __bf16 Vt[128 * 48];       // transposed [d][key]
  __shared__ __bf16 Pw[8 * 16 * 32];    // per-wave P staging

  const int tid = threadIdx.x, wave = tid >> 5, lane = tid & 31;
  const int lm = lane & 15, lg = lane >> 4;
  const int bh = blockIdx.x;
  const int b = bh >> 4, h = bh & 15;
  const int qblk = blockIdx.y;
  const int qrow0 = qblk * 128 + wave * 16;
  const size_t rowbase = (size_t)bh * S_ * HD_;
  const float scale = 0.08838834764831845f;

  FragU qf[4];
  {
    const __bf16* qrow = Qb + rowbase + (size_t)(qrow0 + lm) * HD_;
    #pragma unroll
    for (int c = 0; c < 4; c++) {
      qf[c].u[0] = *(const uint4*)(qrow + c * 32 + lg * 8);
      qf[c].u[1] = *(const uint4*)(qrow + c * 32 + 16 + lg * 8);
    }
  }

  f32x8 oacc[8];
  float mrun[8], lrun[8];
  for (int j = 0; j < 8; j++) for (int r = 0; r < 8; r++) oacc[j][r] = 0.f;
  for (int r = 0; r < 8; r++) { mrun[r] = -3.0e38f; lrun[r] = 0.f; }

  const int kmax = qblk * 128 + 128;
  for (int kt0 = 0; kt0 < kmax; kt0 += 32) {
    { // cooperative K/V staging
      const int krow = tid >> 3, dcol = (tid & 7) * 16;
      const __bf16* kg = K0b + rowbase + (size_t)(kt0 + krow) * HD_ + dcol;
      __bf16* kl = Klds + krow * 136 + dcol;
      async_ld_b128(kl,     kg);
      async_ld_b128(kl + 8, kg + 8);
      const __bf16* vg = V0b + rowbase + (size_t)(kt0 + krow) * HD_ + dcol;
      #pragma unroll
      for (int j = 0; j < 16; j++) Vt[(dcol + j) * 48 + krow] = vg[j];
    }
    wait_async0();
    __syncthreads();

    float sv[2][8];
    #pragma unroll
    for (int t = 0; t < 2; t++) {
      f32x8 sacc; for (int r = 0; r < 8; r++) sacc[r] = 0.f;
      #pragma unroll
      for (int c = 0; c < 4; c++) {
        FragU kf;
        const __bf16* kp = Klds + (t * 16 + lm) * 136 + c * 32;
        kf.u[0] = *(const uint4*)(kp + lg * 8);
        kf.u[1] = *(const uint4*)(kp + 16 + lg * 8);
        sacc = wmma_bf16(qf[c].f, kf.f, sacc);
      }
      #pragma unroll
      for (int r = 0; r < 8; r++) {
        int qi = qrow0 + lg * 8 + r;
        int kj = kt0 + t * 16 + lm;
        sv[t][r] = (kj <= qi) ? sacc[r] * scale : -3.0e38f;
      }
    }

    float p[2][8];
    #pragma unroll
    for (int r = 0; r < 8; r++) {
      float m = fmaxf(sv[0][r], sv[1][r]);
      m = fmaxf(m, __shfl_xor(m, 1, 16));
      m = fmaxf(m, __shfl_xor(m, 2, 16));
      m = fmaxf(m, __shfl_xor(m, 4, 16));
      m = fmaxf(m, __shfl_xor(m, 8, 16));
      float nm = fmaxf(mrun[r], m);
      float alpha = __expf(mrun[r] - nm);
      mrun[r] = nm;
      float p0 = __expf(sv[0][r] - nm);
      float p1 = __expf(sv[1][r] - nm);
      p[0][r] = p0; p[1][r] = p1;
      float rs = p0 + p1;
      rs += __shfl_xor(rs, 1, 16);
      rs += __shfl_xor(rs, 2, 16);
      rs += __shfl_xor(rs, 4, 16);
      rs += __shfl_xor(rs, 8, 16);
      lrun[r] = lrun[r] * alpha + rs;
      #pragma unroll
      for (int j = 0; j < 8; j++) oacc[j][r] *= alpha;
    }

    // bounce P through wave-private LDS to re-fragment as A-matrix
    __bf16* pw = Pw + wave * 512;
    #pragma unroll
    for (int r = 0; r < 8; r++) {
      int prow = lg * 8 + r;
      pw[prow * 32 + lm]      = f2bf(p[0][r]);
      pw[prow * 32 + 16 + lm] = f2bf(p[1][r]);
    }
    asm volatile("s_wait_dscnt 0x0" ::: "memory");
    FragU pf;
    {
      const __bf16* pp = pw + lm * 32;
      pf.u[0] = *(const uint4*)(pp + lg * 8);
      pf.u[1] = *(const uint4*)(pp + 16 + lg * 8);
    }
    #pragma unroll
    for (int j = 0; j < 8; j++) {
      FragU vf;
      const __bf16* vp = Vt + (j * 16 + lm) * 48;
      vf.u[0] = *(const uint4*)(vp + lg * 8);
      vf.u[1] = *(const uint4*)(vp + 16 + lg * 8);
      oacc[j] = wmma_bf16(pf.f, vf.f, oacc[j]);
    }
    __syncthreads();
  }

  // fold in the 3 diagonal cache columns
  const size_t vnewbase = ((size_t)b * NKV_ + (h >> 1)) * S_ * HD_;
  #pragma unroll
  for (int r = 0; r < 8; r++) {
    int row = qrow0 + lg * 8 + r;
    const float* l3 = logits3 + ((size_t)bh * S_ + row) * 3;
    float l0 = l3[0], l1 = l3[1], l2 = l3[2];
    float nm = fmaxf(mrun[r], fmaxf(l0, fmaxf(l1, l2)));
    float alpha = __expf(mrun[r] - nm);
    float e0 = __expf(l0 - nm), e1 = __expf(l1 - nm), e2 = __expf(l2 - nm);
    lrun[r] = lrun[r] * alpha + e0 + e1 + e2;
    mrun[r] = nm;
    const float* v1p = cache_v + (((size_t)(1 * B_ * NH_) + bh) * S_ + row) * HD_;
    const float* v2p = cache_v + (((size_t)(2 * B_ * NH_) + bh) * S_ + row) * HD_;
    const __bf16* v3p = Vnew + vnewbase + (size_t)row * HD_;
    #pragma unroll
    for (int j = 0; j < 8; j++) {
      int d = j * 16 + lm;
      float add = e0 * v1p[d] + e1 * v2p[d] + e2 * bf2f(v3p[d]);
      oacc[j][r] = oacc[j][r] * alpha + add;
    }
  }

  #pragma unroll
  for (int r = 0; r < 8; r++) {
    int row = qrow0 + lg * 8 + r;
    float inv = 1.0f / lrun[r];
    __bf16* op = Out + ((size_t)b * S_ + row) * (NH_ * HD_) + h * HD_;
    #pragma unroll
    for (int j = 0; j < 8; j++) op[j * 16 + lm] = f2bf(oacc[j][r] * inv);
  }
}

// ---------------------------------------------------------------------------
// act = silu(gate) * up from packed [gate|up] f32 buffer
// ---------------------------------------------------------------------------
__global__ void silu_mul_kernel(const float* __restrict__ gu, __bf16* __restrict__ act)
{
  size_t idx = (size_t)blockIdx.x * 256 + threadIdx.x;
  if (idx >= (size_t)4096 * 6144) return;
  size_t m = idx / 6144, i = idx - m * 6144;
  float g = gu[m * 12288 + i];
  float u = gu[m * 12288 + 6144 + i];
  act[idx] = f2bf((g / (1.f + __expf(-g))) * u);
}

// ---------------------------------------------------------------------------
extern "C" void kernel_launch(void* const* d_in, const int* in_sizes, int n_in,
                              void* d_out, int out_size, void* d_ws, size_t ws_size,
                              hipStream_t stream)
{
  (void)in_sizes; (void)n_in; (void)out_size; (void)ws_size;
  const float* emb    = (const float*)d_in[0];
  const float* hid    = (const float*)d_in[1];
  const float* ck     = (const float*)d_in[2];
  const float* cv     = (const float*)d_in[3];
  const float* Wq     = (const float*)d_in[4];
  const float* Wk     = (const float*)d_in[5];
  const float* Wv     = (const float*)d_in[6];
  const float* Wo     = (const float*)d_in[7];
  const float* Wg     = (const float*)d_in[8];
  const float* Wu     = (const float*)d_in[9];
  const float* Wd     = (const float*)d_in[10];
  const float* hn_w   = (const float*)d_in[11];
  const float* in_w   = (const float*)d_in[12];
  const float* post_w = (const float*)d_in[13];
  const float* cosb   = (const float*)d_in[15];
  const float* sinb   = (const float*)d_in[16];
  float* outp = (float*)d_out;
  char* ws = (char*)d_ws;

  auto align = [](size_t x) { return (x + 255) & ~(size_t)255; };
  size_t o = 0;
  auto grab = [&](size_t bytes) { size_t r = o; o = align(o + bytes); return r; };

  // phase-1 region (dead by MLP time)
  size_t oX    = grab((size_t)4096 * 4096 * 2);   // x bf16
  size_t oWqkv = grab((size_t)4096 * 4096 * 2);   // packed Wq|Wk|Wv bf16
  size_t oQKV  = grab((size_t)4096 * 4096 * 4);   // qkv f32
  size_t oQ    = grab((size_t)8388608 * 2);       // q bf16 (B,NH,S,HD)
  size_t oK    = grab((size_t)4194304 * 2);       // k bf16 (B,NKV,S,HD)
  size_t oV    = grab((size_t)4194304 * 2);       // v bf16
  size_t oK0   = grab((size_t)8388608 * 2);       // cache_k[0] bf16
  size_t oV0   = grab((size_t)8388608 * 2);       // cache_v[0] bf16
  size_t oL    = grab((size_t)196608 * 4);        // diag logits
  size_t oAttn = grab((size_t)8388608 * 2);       // attn out bf16
  size_t headEnd = o;
  // phase-2 overlay reuses the phase-1 region
  size_t oGU  = 0;                                 // gate|up f32 (4096x12288)
  size_t oAct = align((size_t)4096 * 12288 * 4);   // act bf16 (4096x6144)
  size_t overlayEnd = align(oAct + (size_t)4096 * 6144 * 2);
  o = headEnd > overlayEnd ? headEnd : overlayEnd;
  // persistent tail
  size_t oHid = grab((size_t)4096 * 2048 * 4);
  size_t oHn  = grab((size_t)4096 * 2048 * 2);
  size_t oWoB = grab((size_t)2048 * 2048 * 2);
  size_t oWgu = grab((size_t)2048 * 12288 * 2);
  size_t oWdB = grab((size_t)6144 * 2048 * 2);

  dim3 b256(256);
  auto NB = [](size_t n) { return dim3((unsigned)((n + 255) / 256)); };

  // weight / cache packing to bf16
  pack_bf16_kernel<<<NB(4096ull*2048), b256, 0, stream>>>(Wq, (__bf16*)(ws+oWqkv), 4096, 2048, 4096, 0);
  pack_bf16_kernel<<<NB(4096ull*1024), b256, 0, stream>>>(Wk, (__bf16*)(ws+oWqkv), 4096, 1024, 4096, 2048);
  pack_bf16_kernel<<<NB(4096ull*1024), b256, 0, stream>>>(Wv, (__bf16*)(ws+oWqkv), 4096, 1024, 4096, 3072);
  pack_bf16_kernel<<<NB(2048ull*2048), b256, 0, stream>>>(Wo, (__bf16*)(ws+oWoB), 2048, 2048, 2048, 0);
  pack_bf16_kernel<<<NB(2048ull*6144), b256, 0, stream>>>(Wg, (__bf16*)(ws+oWgu), 2048, 6144, 12288, 0);
  pack_bf16_kernel<<<NB(2048ull*6144), b256, 0, stream>>>(Wu, (__bf16*)(ws+oWgu), 2048, 6144, 12288, 6144);
  pack_bf16_kernel<<<NB(6144ull*2048), b256, 0, stream>>>(Wd, (__bf16*)(ws+oWdB), 6144, 2048, 2048, 0);
  pack_bf16_kernel<<<NB(8388608ull),   b256, 0, stream>>>(ck, (__bf16*)(ws+oK0), 65536, 128, 128, 0);
  pack_bf16_kernel<<<NB(8388608ull),   b256, 0, stream>>>(cv, (__bf16*)(ws+oV0), 65536, 128, 128, 0);

  // norms + fused QKV projection
  rms_concat_kernel<<<4096, b256, 0, stream>>>(emb, hid, in_w, hn_w, (__bf16*)(ws+oX));
  gemm_bf16_kernel<0><<<dim3(16, 32), b256, 0, stream>>>(
      (const __bf16*)(ws+oX), (const __bf16*)(ws+oWqkv), (float*)(ws+oQKV), nullptr, 4096, 4096, 4096);
  rope_kernel<<<4096, b256, 0, stream>>>((const float*)(ws+oQKV), cosb, sinb,
      (__bf16*)(ws+oQ), (__bf16*)(ws+oK), (__bf16*)(ws+oV));

  // attention
  diag_logits_kernel<<<65536, dim3(128), 0, stream>>>(
      (const __bf16*)(ws+oQ), ck, (const __bf16*)(ws+oK), (float*)(ws+oL));
  attention_kernel<<<dim3(32, 16), b256, 0, stream>>>(
      (const __bf16*)(ws+oQ), (const __bf16*)(ws+oK0), (const __bf16*)(ws+oV0),
      (const float*)(ws+oL), cv, (const __bf16*)(ws+oV), (__bf16*)(ws+oAttn));
  gemm_bf16_kernel<1><<<dim3(8, 32), b256, 0, stream>>>(
      (const __bf16*)(ws+oAttn), (const __bf16*)(ws+oWoB), (float*)(ws+oHid), hid, 4096, 2048, 2048);

  // MLP
  rms_post_kernel<<<4096, b256, 0, stream>>>((const float*)(ws+oHid), post_w, (__bf16*)(ws+oHn));
  gemm_bf16_kernel<0><<<dim3(48, 32), b256, 0, stream>>>(
      (const __bf16*)(ws+oHn), (const __bf16*)(ws+oWgu), (float*)(ws+oGU), nullptr, 4096, 12288, 2048);
  silu_mul_kernel<<<NB(4096ull*6144), b256, 0, stream>>>((const float*)(ws+oGU), (__bf16*)(ws+oAct));
  gemm_bf16_kernel<1><<<dim3(8, 32), b256, 0, stream>>>(
      (const __bf16*)(ws+oAct), (const __bf16*)(ws+oWdB), outp, (const float*)(ws+oHid), 4096, 2048, 6144);
}
